// Attention_Text_42391327212018
// MI455X (gfx1250) — matrix operand order repaired and checked by reading, not compiled, not run on hardware
//
#include <hip/hip_runtime.h>
#include <hip/hip_bf16.h>
#include <stdint.h>

typedef __bf16 bf16;
typedef __attribute__((ext_vector_type(16))) __bf16 v16bf;
typedef __attribute__((ext_vector_type(8)))  __bf16 v8bf;
typedef __attribute__((ext_vector_type(4)))  __bf16 v4bf;
typedef __attribute__((ext_vector_type(8)))  float   v8f;
typedef __attribute__((ext_vector_type(4)))  unsigned int u32x4;
typedef __attribute__((ext_vector_type(8)))  int i32x8;
typedef __attribute__((ext_vector_type(4)))  int i32x4;

#define B_SZ 8
#define NVQ  1024
#define NTK  1024
#define DV   2048
#define DTD  1024

#define BM 128
#define BN 128
#define BK 32
#define LDK 48   // padded K-stride in LDS (elements); 96B rows, 16B aligned

__device__ __forceinline__ bf16 f2bf(float f) {
  union { float f; unsigned u; } a; a.f = f;
  unsigned r = a.u + 0x7FFFu + ((a.u >> 16) & 1u);   // round-to-nearest-even
  union { unsigned short s; bf16 b; } o; o.s = (unsigned short)(r >> 16);
  return o.b;
}

// A-fragment 16x32 bf16: lane half h holds K in {8h..8h+7} U {16+8h..23+8h}
__device__ __forceinline__ v16bf frag_a(const bf16* rowp, int h) {
  v8bf lo = *(const v8bf*)(rowp + 8 * h);
  v8bf hi = *(const v8bf*)(rowp + 16 + 8 * h);
  v16bf r;
#pragma unroll
  for (int i = 0; i < 8; ++i) { r[i] = lo[i]; r[i + 8] = hi[i]; }
  return r;
}

// B-fragment 32x16 bf16: lane half h holds K in {8h..8h+7} U {16+8h..23+8h}
__device__ __forceinline__ v16bf frag_b(const bf16* colp, int h) {
  v8bf lo = *(const v8bf*)(colp + 8 * h);
  v8bf hi = *(const v8bf*)(colp + 16 + 8 * h);
  v16bf r;
#pragma unroll
  for (int i = 0; i < 8; ++i) { r[i] = lo[i]; r[i + 8] = hi[i]; }
  return r;
}

// TDM 2D tile load: global (row-major, bf16) -> LDS with pad to LDK=48 rows.
// pad: every 16 DWORDs (32 bf16 = one 64B tile row) insert 8 DWORDs (32B) -> 96B row stride.
__device__ __forceinline__ void tdm_load_2d(unsigned lds_off, const bf16* gptr,
                                            unsigned tensor_w, unsigned tensor_h,
                                            unsigned tile_w, unsigned tile_h,
                                            unsigned row_stride) {
  unsigned long long ga = (unsigned long long)(uintptr_t)gptr;
  u32x4 g0;
  g0[0] = 1u;                                               // count=1, user desc
  g0[1] = lds_off;                                          // LDS byte address
  g0[2] = (unsigned)ga;                                     // global addr lo
  g0[3] = (unsigned)((ga >> 32) & 0x1FFFFFFu) | (2u << 30); // addr hi | type=2
  i32x8 g1;
  g1[0] = (int)((1u << 16) | (1u << 20) | (3u << 22) | (7u << 25)); // 2B elems, pad en, every 16DW add 8DW
  g1[1] = (int)(tensor_w << 16);
  g1[2] = (int)((tensor_w >> 16) | ((tensor_h & 0xFFFFu) << 16));
  g1[3] = (int)((tensor_h >> 16) | (tile_w << 16));
  g1[4] = (int)tile_h;                                      // tile_dim1; tile_dim2=0
  g1[5] = (int)row_stride;                                  // dim0 stride lo (elements)
  g1[6] = 0;
  g1[7] = 0;
  i32x4 z4 = {0, 0, 0, 0};
  i32x8 z8 = {0, 0, 0, 0, 0, 0, 0, 0};
  __builtin_amdgcn_tensor_load_to_lds(g0, g1, z4, z4, z8, 0);
}

// ---------------- Kernel 0: text fp32 -> bf16 --------------------------------
__global__ __launch_bounds__(256) void k_cvt(const float* __restrict__ x,
                                             bf16* __restrict__ y) {
  const size_t i = ((size_t)blockIdx.x * 256 + threadIdx.x) * 4;
  const float4 v = *(const float4*)(x + i);
  v4bf o; o[0] = f2bf(v.x); o[1] = f2bf(v.y); o[2] = f2bf(v.z); o[3] = f2bf(v.w);
  *(v4bf*)(y + i) = o;
}

// ---------------- Kernel 1: q = visual @ W^T + bias  (M=B*NV, N=DT, K=DV) ----
__global__ __launch_bounds__(256) void k_qproj(const float* __restrict__ vis,
                                               const float* __restrict__ W,
                                               const float* __restrict__ bias,
                                               bf16* __restrict__ q) {
  __shared__ bf16 As[BM * LDK];
  __shared__ bf16 Bs[BN * LDK];
  const int tid = threadIdx.x;
  const int lane = tid & 31;
  const int w  = tid >> 5;
  const int wm = w & 3;
  const int wn = w >> 2;
  const int r  = lane & 15;
  const int h  = lane >> 4;
  const int M0 = blockIdx.y * BM;
  const int N0 = blockIdx.x * BN;

  v8f acc[2][4] = {};

  const int a_m = tid >> 3;            // 0..31
  const int a_k = (tid & 7) * 4;       // fp32x4 per thread

  for (int k0 = 0; k0 < DV; k0 += BK) {
#pragma unroll
    for (int p = 0; p < 4; ++p) {      // A: 128 rows x 32 k
      int m = a_m + p * 32;
      const float4 v = *(const float4*)(vis + (size_t)(M0 + m) * DV + k0 + a_k);
      v4bf o; o[0] = f2bf(v.x); o[1] = f2bf(v.y); o[2] = f2bf(v.z); o[3] = f2bf(v.w);
      *(v4bf*)(As + m * LDK + a_k) = o;
    }
#pragma unroll
    for (int p = 0; p < 4; ++p) {      // B: Bs[n][k] = W[N0+n][k0+k]
      int n = a_m + p * 32;
      const float4 v = *(const float4*)(W + (size_t)(N0 + n) * DV + k0 + a_k);
      v4bf o; o[0] = f2bf(v.x); o[1] = f2bf(v.y); o[2] = f2bf(v.z); o[3] = f2bf(v.w);
      *(v4bf*)(Bs + n * LDK + a_k) = o;
    }
    __syncthreads();

    v16bf af[2], bfr[4];
#pragma unroll
    for (int mi = 0; mi < 2; ++mi) af[mi]  = frag_a(As + (wm * 32 + mi * 16 + r) * LDK, h);
#pragma unroll
    for (int ni = 0; ni < 4; ++ni) bfr[ni] = frag_b(Bs + (wn * 64 + ni * 16 + r) * LDK, h);
#pragma unroll
    for (int mi = 0; mi < 2; ++mi)
#pragma unroll
      for (int ni = 0; ni < 4; ++ni)
        acc[mi][ni] = __builtin_amdgcn_wmma_f32_16x16x32_bf16(
            false, af[mi], false, bfr[ni], (short)0, acc[mi][ni], false, false);
    __syncthreads();
  }

#pragma unroll
  for (int ni = 0; ni < 4; ++ni) {
    const int n = N0 + wn * 64 + ni * 16 + r;
    const float bv = bias[n];
#pragma unroll
    for (int mi = 0; mi < 2; ++mi)
#pragma unroll
      for (int v = 0; v < 8; ++v) {
        const int m = M0 + wm * 32 + mi * 16 + v + 8 * h;
        q[(size_t)m * DTD + n] = f2bf(acc[mi][ni][v] + bv);
      }
  }
}

// ---------------- Kernel 2: S = q @ text^T  (per batch: M=NV, N=NT, K=DT) ----
// Both tiles are bf16 in memory -> staged by the Tensor Data Mover.
__global__ __launch_bounds__(256) void k_scores(const bf16* __restrict__ q,
                                                const bf16* __restrict__ tx,
                                                float* __restrict__ S) {
  __shared__ bf16 As[BM * LDK];
  __shared__ bf16 Bs[BN * LDK];
  const int tid = threadIdx.x;
  const int lane = tid & 31;
  const int w  = tid >> 5;
  const int wm = w & 3;
  const int wn = w >> 2;
  const int r  = lane & 15;
  const int h  = lane >> 4;
  const int b  = blockIdx.z;
  const int M0 = blockIdx.y * BM;      // over NV
  const int N0 = blockIdx.x * BN;      // over NT
  const bf16* qb = q  + (size_t)b * NVQ * DTD;
  const bf16* tb = tx + (size_t)b * NTK * DTD;
  const unsigned lds_a = (unsigned)(uintptr_t)As;
  const unsigned lds_b = (unsigned)(uintptr_t)Bs;

  v8f acc[2][4] = {};

  for (int k0 = 0; k0 < DTD; k0 += BK) {
    if (w == 0) {
      tdm_load_2d(lds_a, qb + (size_t)M0 * DTD + k0, DTD, NVQ, BK, BM, DTD);
      tdm_load_2d(lds_b, tb + (size_t)N0 * DTD + k0, DTD, NTK, BK, BN, DTD);
      __builtin_amdgcn_s_wait_tensorcnt(0);
    }
    __syncthreads();

    v16bf af[2], bfr[4];
#pragma unroll
    for (int mi = 0; mi < 2; ++mi) af[mi]  = frag_a(As + (wm * 32 + mi * 16 + r) * LDK, h);
#pragma unroll
    for (int ni = 0; ni < 4; ++ni) bfr[ni] = frag_b(Bs + (wn * 64 + ni * 16 + r) * LDK, h);
#pragma unroll
    for (int mi = 0; mi < 2; ++mi)
#pragma unroll
      for (int ni = 0; ni < 4; ++ni)
        acc[mi][ni] = __builtin_amdgcn_wmma_f32_16x16x32_bf16(
            false, af[mi], false, bfr[ni], (short)0, acc[mi][ni], false, false);
    __syncthreads();
  }

#pragma unroll
  for (int ni = 0; ni < 4; ++ni) {
    const int n = N0 + wn * 64 + ni * 16 + r;
#pragma unroll
    for (int mi = 0; mi < 2; ++mi)
#pragma unroll
      for (int v = 0; v < 8; ++v) {
        const int m = M0 + wm * 32 + mi * 16 + v + 8 * h;
        S[((size_t)b * NVQ + m) * NTK + n] = acc[mi][ni][v];
      }
  }
}

// ---------------- Kernel 3: row softmax over NT, output bf16 P ---------------
__global__ __launch_bounds__(256) void k_softmax(const float* __restrict__ S,
                                                 bf16* __restrict__ P) {
  __shared__ float red[256];
  const size_t row = blockIdx.x;
  const int tid = threadIdx.x;
  const float4 x = *(const float4*)(S + row * NTK + tid * 4);
  float m = fmaxf(fmaxf(x.x, x.y), fmaxf(x.z, x.w));
  red[tid] = m; __syncthreads();
  for (int s = 128; s > 0; s >>= 1) {
    if (tid < s) red[tid] = fmaxf(red[tid], red[tid + s]);
    __syncthreads();
  }
  const float mx = red[0]; __syncthreads();
  const float e0 = __expf(x.x - mx), e1 = __expf(x.y - mx);
  const float e2 = __expf(x.z - mx), e3 = __expf(x.w - mx);
  red[tid] = e0 + e1 + e2 + e3; __syncthreads();
  for (int s = 128; s > 0; s >>= 1) {
    if (tid < s) red[tid] += red[tid + s];
    __syncthreads();
  }
  const float inv = 1.0f / red[0];
  v4bf o; o[0] = f2bf(e0 * inv); o[1] = f2bf(e1 * inv);
          o[2] = f2bf(e2 * inv); o[3] = f2bf(e3 * inv);
  *(v4bf*)(P + row * NTK + tid * 4) = o;
}

// ---------------- Kernel 4: out = P @ text  (per batch: M=NV, N=DT, K=NT) ----
// A tile (P, bf16) staged by TDM; B tile needs transpose -> manual bf16 stage.
__global__ __launch_bounds__(256) void k_av(const bf16* __restrict__ P,
                                            const bf16* __restrict__ tx,
                                            float* __restrict__ out) {
  __shared__ bf16 As[BM * LDK];
  __shared__ bf16 Bs[BN * LDK];
  const int tid = threadIdx.x;
  const int lane = tid & 31;
  const int w  = tid >> 5;
  const int wm = w & 3;
  const int wn = w >> 2;
  const int r  = lane & 15;
  const int h  = lane >> 4;
  const int b  = blockIdx.z;
  const int M0 = blockIdx.y * BM;      // over NV
  const int N0 = blockIdx.x * BN;      // over DT
  const bf16* pb = P  + (size_t)b * NVQ * NTK;
  const bf16* tb = tx + (size_t)b * NTK * DTD;
  const unsigned lds_a = (unsigned)(uintptr_t)As;

  v8f acc[2][4] = {};

  const int bn8 = (tid & 15) * 8;      // 8 consecutive N per thread
  const int bkk = tid >> 4;            // 0..15 K rows per pass

  for (int k0 = 0; k0 < NTK; k0 += BK) {
    if (w == 0)
      tdm_load_2d(lds_a, pb + (size_t)M0 * NTK + k0, NTK, NVQ, BK, BM, NTK);
#pragma unroll
    for (int p = 0; p < 2; ++p) {      // Bs[n][k] = text[k0+k][N0+n] (transposed)
      int k = bkk + p * 16;
      v8bf v = *(const v8bf*)(tb + (size_t)(k0 + k) * DTD + N0 + bn8);
#pragma unroll
      for (int j = 0; j < 8; ++j) Bs[(bn8 + j) * LDK + k] = v[j];
    }
    if (w == 0) __builtin_amdgcn_s_wait_tensorcnt(0);
    __syncthreads();

    v16bf af[2], bfr[4];
#pragma unroll
    for (int mi = 0; mi < 2; ++mi) af[mi]  = frag_a(As + (wm * 32 + mi * 16 + r) * LDK, h);
#pragma unroll
    for (int ni = 0; ni < 4; ++ni) bfr[ni] = frag_b(Bs + (wn * 64 + ni * 16 + r) * LDK, h);
#pragma unroll
    for (int mi = 0; mi < 2; ++mi)
#pragma unroll
      for (int ni = 0; ni < 4; ++ni)
        acc[mi][ni] = __builtin_amdgcn_wmma_f32_16x16x32_bf16(
            false, af[mi], false, bfr[ni], (short)0, acc[mi][ni], false, false);
    __syncthreads();
  }

#pragma unroll
  for (int ni = 0; ni < 4; ++ni) {
    const int n = N0 + wn * 64 + ni * 16 + r;
#pragma unroll
    for (int mi = 0; mi < 2; ++mi)
#pragma unroll
      for (int v = 0; v < 8; ++v) {
        const int m = M0 + wm * 32 + mi * 16 + v + 8 * h;
        out[((size_t)b * NVQ + m) * DTD + n] = acc[mi][ni][v];
      }
  }
}

extern "C" void kernel_launch(void* const* d_in, const int* in_sizes, int n_in,
                              void* d_out, int out_size, void* d_ws, size_t ws_size,
                              hipStream_t stream) {
  (void)in_sizes; (void)n_in; (void)out_size; (void)ws_size;
  const float* vis  = (const float*)d_in[0];
  const float* text = (const float*)d_in[1];
  const float* Ww   = (const float*)d_in[2];
  const float* Wb   = (const float*)d_in[3];
  float* out = (float*)d_out;

  char* ws = (char*)d_ws;
  bf16*  q    = (bf16*)ws;                                  // 16 MB bf16 q
  float* S    = (float*)(ws + (size_t)16 * 1024 * 1024);    // 32 MB f32 scores
  bf16*  P    = (bf16*) (ws + (size_t)48 * 1024 * 1024);    // 16 MB bf16 probs
  bf16*  tb16 = (bf16*) (ws + (size_t)64 * 1024 * 1024);    // 16 MB bf16 text

  dim3 blk(256);
  k_cvt    <<<dim3((B_SZ * NTK * DTD) / (256 * 4)),  blk, 0, stream>>>(text, tb16);
  k_qproj  <<<dim3(DTD / BN, (B_SZ * NVQ) / BM),     blk, 0, stream>>>(vis, Ww, Wb, q);
  k_scores <<<dim3(NTK / BN, NVQ / BM, B_SZ),        blk, 0, stream>>>(q, tb16, S);
  k_softmax<<<dim3(B_SZ * NVQ),                      blk, 0, stream>>>(S, P);
  k_av     <<<dim3(DTD / BN, NVQ / BM, B_SZ),        blk, 0, stream>>>(P, tb16, out);
}
